// QFCModelExtended_65481071407970
// MI455X (gfx1250) — matrix operand order repaired and checked by reading, not compiled, not run on hardware
//
#include <hip/hip_runtime.h>

typedef __attribute__((ext_vector_type(16))) _Float16 v16h;
typedef __attribute__((ext_vector_type(8)))  float    v8f;

#define KDIM   576      // 24*24 pixels per sample
#define KITER  18       // 576 / 32
#define TPB    256      // 8 waves, 256 samples per block

// ws layout in floats
#define PART_OFF 0      // nblocks * 8 partial sums (nblocks <= 512)
#define SC_OFF   4096   // scale[4], shift[4]
#define RZ_OFF   4104   // rz cos/sin per (layer,qubit): 2*4*2 = 16 floats
#define WH_OFF   4128   // WMMA-B-fragment-ordered f16 weights: 18*32*16 halfs = 4608 floats

// ---------------------------------------------------------------------------
// Prep: build fragment-ordered f16 fused pooling+encoding weight matrix, and
// the batch-uniform RZ phase constants.  W[k][n] = enc_w[n][cell(k)]/36 (n<4).
// Fragment order for V_WMMA_F32_16X16X32_F16 B (32x16):
//   flat = ((iter*32 + lane)*16 + j)  <->  n = lane%16, k = iter*32 + 8*(lane/16) + (j&7) + 16*(j/8)
// ---------------------------------------------------------------------------
__global__ void qfc_prep(const float* __restrict__ enc_w,
                         const float* __restrict__ q_params,
                         float* __restrict__ wsf) {
  const int tid = threadIdx.x;
  if (tid < 8) {
    float th = q_params[tid];                  // [layer*4 + q]
    wsf[RZ_OFF + tid * 2 + 0] = cosf(0.5f * th);
    wsf[RZ_OFF + tid * 2 + 1] = sinf(0.5f * th);
  }
  _Float16* wh = (_Float16*)(wsf + WH_OFF);
  for (int t = tid; t < KITER * 32 * 16; t += blockDim.x) {
    const int i = t >> 9;            // /512 : k-iteration
    const int l = (t >> 4) & 31;     // lane
    const int j = t & 15;            // half within lane record
    const int n = l & 15;            // output column
    const int k = i * 32 + ((l >> 4) << 3) + (j & 7) + ((j >> 3) << 4);   // pixel index 0..575
    float v = 0.0f;
    if (n < 4) {
      const int r = k / 24, c = k % 24;
      const int cell = (r / 6) * 4 + (c / 6);     // pooled cell 0..15
      v = enc_w[n * 16 + cell] * (1.0f / 36.0f);
    }
    wh[t] = (_Float16)v;
  }
}

// ---------------------------------------------------------------------------
// Circuit gate helpers (fully unrolled -> pure register ops)
// qubit q acts on bit (3-q) of the 16-dim state index.
// ---------------------------------------------------------------------------
__device__ __forceinline__ void ry_gate(float sr[16], float si[16], const int q,
                                        float c, float s) {
  const int b = 1 << (3 - q);
#pragma unroll
  for (int i = 0; i < 16; ++i) {
    if ((i & b) == 0) {
      const int j = i | b;
      float ar = sr[i], br = sr[j], ai = si[i], bi = si[j];
      sr[i] = c * ar - s * br;  sr[j] = s * ar + c * br;
      si[i] = c * ai - s * bi;  si[j] = s * ai + c * bi;
    }
  }
}

__device__ __forceinline__ void rz_gate(float sr[16], float si[16], const int q,
                                        float pc, float ps) {
  const int b = 1 << (3 - q);
#pragma unroll
  for (int i = 0; i < 16; ++i) {
    float r = sr[i], m = si[i];
    if ((i & b) == 0) { sr[i] = r * pc + m * ps;  si[i] = m * pc - r * ps; }
    else              { sr[i] = r * pc - m * ps;  si[i] = m * pc + r * ps; }
  }
}

__device__ __forceinline__ void cnot_gate(float sr[16], float si[16],
                                          const int c, const int t) {
  const int cb = 1 << (3 - c), tb = 1 << (3 - t);
#pragma unroll
  for (int i = 0; i < 16; ++i) {
    if ((i & cb) != 0 && (i & tb) == 0) {
      const int j = i | tb;
      float tr = sr[i]; sr[i] = sr[j]; sr[j] = tr;
      float ti = si[i]; si[i] = si[j]; si[j] = ti;
    }
  }
}

// ---------------------------------------------------------------------------
// Main fused kernel: WMMA encode (8 waves x 2 tiles x 18 k-steps) -> per-thread
// quantum circuit -> raw z to d_out -> deterministic per-block sum/sumsq.
// ---------------------------------------------------------------------------
__global__ __launch_bounds__(TPB) void qfc_main(const float* __restrict__ x,
                                                const float* __restrict__ enc_b,
                                                const float* __restrict__ wsf,
                                                float* __restrict__ out,
                                                float* __restrict__ partial) {
  __shared__ __align__(32) _Float16 ldsB[KITER * 32 * 16];   // 18 KB weights
  __shared__ float encS[TPB][4];                             // encoded angles
  __shared__ float wred[8][8];                               // wave partials

  const int tid  = threadIdx.x;
  const int wv   = tid >> 5;
  const int lane = tid & 31;
  const int hi   = lane >> 4;        // which half of the wave
  const int lm   = lane & 15;        // M-row (A) / N-col (B,C) selector

  // Stage fragment-ordered weights into LDS (coalesced 16B copies).
  {
    const uint4* src = (const uint4*)(wsf + WH_OFF);
    uint4* dst = (uint4*)ldsB;
    for (int j = tid; j < (KITER * 32 * 16 * 2) / 16; j += TPB) dst[j] = src[j];
  }
  __syncthreads();

  // ---------------- WMMA encoding GEMM: two 16-row tiles per wave ----------
  const int blockBase = blockIdx.x * TPB;
  const int waveBase  = blockBase + wv * 32;
  const float* row0 = x + (size_t)(waveBase + lm) * KDIM;        // tile 0
  const float* row1 = x + (size_t)(waveBase + 16 + lm) * KDIM;   // tile 1

  v8f acc0 = {}, acc1 = {};
#pragma unroll 2
  for (int i = 0; i < KITER; ++i) {
    const int kb = i * 32;
    // B fragment: 32 contiguous bytes per lane from LDS
    v16h bfrag = *(const v16h*)&ldsB[(i * 32 + lane) * 16];
    // A fragments: lane holds row lm, K = kb+8*hi+[0..7] and +16..[0..7]
    const float* p0 = row0 + kb + hi * 8;
    const float* p1 = row1 + kb + hi * 8;
    float4 a00 = *(const float4*)(p0);      float4 a01 = *(const float4*)(p0 + 4);
    float4 a02 = *(const float4*)(p0 + 16); float4 a03 = *(const float4*)(p0 + 20);
    float4 b00 = *(const float4*)(p1);      float4 b01 = *(const float4*)(p1 + 4);
    float4 b02 = *(const float4*)(p1 + 16); float4 b03 = *(const float4*)(p1 + 20);
    if (i + 1 < KITER) {                    // prefetch next k-chunk
      __builtin_prefetch(p0 + 32, 0, 0);
      __builtin_prefetch(p1 + 32, 0, 0);
    }
    v16h a0, a1;
    a0[0]=(_Float16)a00.x; a0[1]=(_Float16)a00.y; a0[2]=(_Float16)a00.z; a0[3]=(_Float16)a00.w;
    a0[4]=(_Float16)a01.x; a0[5]=(_Float16)a01.y; a0[6]=(_Float16)a01.z; a0[7]=(_Float16)a01.w;
    a0[8]=(_Float16)a02.x; a0[9]=(_Float16)a02.y; a0[10]=(_Float16)a02.z; a0[11]=(_Float16)a02.w;
    a0[12]=(_Float16)a03.x; a0[13]=(_Float16)a03.y; a0[14]=(_Float16)a03.z; a0[15]=(_Float16)a03.w;
    a1[0]=(_Float16)b00.x; a1[1]=(_Float16)b00.y; a1[2]=(_Float16)b00.z; a1[3]=(_Float16)b00.w;
    a1[4]=(_Float16)b01.x; a1[5]=(_Float16)b01.y; a1[6]=(_Float16)b01.z; a1[7]=(_Float16)b01.w;
    a1[8]=(_Float16)b02.x; a1[9]=(_Float16)b02.y; a1[10]=(_Float16)b02.z; a1[11]=(_Float16)b02.w;
    a1[12]=(_Float16)b03.x; a1[13]=(_Float16)b03.y; a1[14]=(_Float16)b03.z; a1[15]=(_Float16)b03.w;

    acc0 = __builtin_amdgcn_wmma_f32_16x16x32_f16(false, a0, false, bfrag,
                                                  (short)0, acc0, false, false);
    acc1 = __builtin_amdgcn_wmma_f32_16x16x32_f16(false, a1, false, bfrag,
                                                  (short)0, acc1, false, false);
  }

  // Transpose needed D columns (0..3) to per-sample layout in LDS, add bias.
  if (lm < 4) {
    const float bq = enc_b[lm];
    const float* d0 = (const float*)&acc0;
    const float* d1 = (const float*)&acc1;
#pragma unroll
    for (int r = 0; r < 8; ++r) {
      encS[wv * 32 + r + 8 * hi][lm]      = d0[r] + bq;   // D: vgpr r = row r(+8)
      encS[wv * 32 + 16 + r + 8 * hi][lm] = d1[r] + bq;
    }
  }
  __syncthreads();

  // ---------------- per-sample quantum circuit (one thread = one sample) ---
  float rzc[8], rzs[8];
#pragma unroll
  for (int k = 0; k < 8; ++k) {
    rzc[k] = wsf[RZ_OFF + k * 2 + 0];
    rzs[k] = wsf[RZ_OFF + k * 2 + 1];
  }
  float4 th = *(const float4*)&encS[tid][0];
  float thv[4] = {th.x, th.y, th.z, th.w};
  float cy[4], sy[4];
#pragma unroll
  for (int q = 0; q < 4; ++q) {
    cy[q] = __cosf(0.5f * thv[q]);
    sy[q] = __sinf(0.5f * thv[q]);
  }

  float sr[16], si[16];
#pragma unroll
  for (int i = 0; i < 16; ++i) { sr[i] = 0.0f; si[i] = 0.0f; }
  sr[0] = 1.0f;

#pragma unroll
  for (int layer = 0; layer < 2; ++layer) {
#pragma unroll
    for (int q = 0; q < 4; ++q) {
      ry_gate(sr, si, q, cy[q], sy[q]);
      rz_gate(sr, si, q, rzc[layer * 4 + q], rzs[layer * 4 + q]);
    }
    cnot_gate(sr, si, 0, 1);
    cnot_gate(sr, si, 1, 2);
    cnot_gate(sr, si, 2, 3);
    cnot_gate(sr, si, 3, 0);
  }

  float z[4] = {0.f, 0.f, 0.f, 0.f};
#pragma unroll
  for (int i = 0; i < 16; ++i) {
    float p = sr[i] * sr[i] + si[i] * si[i];
#pragma unroll
    for (int q = 0; q < 4; ++q)
      z[q] += ((i >> (3 - q)) & 1) ? -p : p;
  }

  // raw z -> d_out (normalized in place later)
  *(float4*)&out[(size_t)(blockBase + tid) * 4] = make_float4(z[0], z[1], z[2], z[3]);

  // ---------------- deterministic block reduction of sum / sumsq ----------
  float acc[8];
#pragma unroll
  for (int q = 0; q < 4; ++q) { acc[q] = z[q]; acc[4 + q] = z[q] * z[q]; }
#pragma unroll
  for (int m = 16; m >= 1; m >>= 1) {
#pragma unroll
    for (int q = 0; q < 8; ++q) acc[q] += __shfl_xor(acc[q], m, 32);
  }
  if (lane == 0) {
#pragma unroll
    for (int q = 0; q < 8; ++q) wred[wv][q] = acc[q];
  }
  __syncthreads();
  if (tid < 8) {
    float s = 0.0f;
#pragma unroll
    for (int w2 = 0; w2 < 8; ++w2) s += wred[w2][tid];
    partial[blockIdx.x * 8 + tid] = s;
  }
}

// ---------------------------------------------------------------------------
// Fold per-block partials -> batchnorm scale/shift (single block, fixed order)
// ---------------------------------------------------------------------------
__global__ void qfc_reduce(const float* __restrict__ bn_w,
                           const float* __restrict__ bn_b,
                           float* __restrict__ wsf, int nblocks, float invB) {
  __shared__ float sums[8];
  const int t = threadIdx.x;
  if (t < 8) {
    float s = 0.0f;
    for (int b = 0; b < nblocks; ++b) s += wsf[PART_OFF + b * 8 + t];
    sums[t] = s;
  }
  __syncthreads();
  if (t < 4) {
    float mean  = sums[t] * invB;
    float var   = sums[4 + t] * invB - mean * mean;
    float scale = bn_w[t] * rsqrtf(var + 1e-5f);
    wsf[SC_OFF + t]     = scale;
    wsf[SC_OFF + 4 + t] = bn_b[t] - mean * scale;
  }
}

// ---------------------------------------------------------------------------
// In-place normalize: one float4 (= one sample, cols 0..3) per thread.
// ---------------------------------------------------------------------------
__global__ __launch_bounds__(TPB) void qfc_norm(float* __restrict__ out,
                                                const float* __restrict__ wsf,
                                                int nsamp) {
  const int i = blockIdx.x * TPB + threadIdx.x;
  if (i < nsamp) {
    float4 sc = *(const float4*)&wsf[SC_OFF];
    float4 sh = *(const float4*)&wsf[SC_OFF + 4];
    float4 v = ((float4*)out)[i];
    v.x = v.x * sc.x + sh.x;
    v.y = v.y * sc.y + sh.y;
    v.z = v.z * sc.z + sh.z;
    v.w = v.w * sc.w + sh.w;
    ((float4*)out)[i] = v;
  }
}

extern "C" void kernel_launch(void* const* d_in, const int* in_sizes, int n_in,
                              void* d_out, int out_size, void* d_ws, size_t ws_size,
                              hipStream_t stream) {
  const float* x        = (const float*)d_in[0];   // (B,1,24,24)
  const float* enc_w    = (const float*)d_in[1];   // (4,16)
  const float* enc_b    = (const float*)d_in[2];   // (4,)
  const float* q_params = (const float*)d_in[3];   // (2,4)
  const float* bn_w     = (const float*)d_in[4];   // (4,)
  const float* bn_b     = (const float*)d_in[5];   // (4,)
  float* out = (float*)d_out;
  float* wsf = (float*)d_ws;

  const int B = in_sizes[0] / KDIM;     // 131072
  const int nblocks = B / TPB;          // 512

  qfc_prep<<<1, 256, 0, stream>>>(enc_w, q_params, wsf);
  qfc_main<<<nblocks, TPB, 0, stream>>>(x, enc_b, wsf, out, wsf + PART_OFF);
  qfc_reduce<<<1, 32, 0, stream>>>(bn_w, bn_b, wsf, nblocks, 1.0f / (float)B);
  qfc_norm<<<(B + TPB - 1) / TPB, TPB, 0, stream>>>(out, wsf, B);
}
